// CrossAttention_50070728737309
// MI455X (gfx1250) — matrix operand order repaired and checked by reading, not compiled, not run on hardware
//
#include <hip/hip_runtime.h>

// ---------------------------------------------------------------------------
// CrossAttention for MI455X (gfx1250, wave32, WMMA f32_16x16x32_f16)
// All WMMA operands are 128-bit vector loads (16B aligned by construction).
// Attention stages K/V tiles through LDS (double-buffered, block-shared).
// ---------------------------------------------------------------------------
typedef _Float16 half_t;
typedef __attribute__((ext_vector_type(16))) _Float16 v16h;
typedef __attribute__((ext_vector_type(8)))  _Float16 v8h;
typedef __attribute__((ext_vector_type(8)))  float    v8f;

#define B_SZ   4
#define CQ     256
#define CKV    512
#define DIM    128
#define HEADS  4
#define HD     32
#define NTOK   2304                  // 48*48
#define NQT    (NTOK/16)             // 144 query tiles
#define NKT    (NTOK/32)             // 72 key tiles of 32
#define EPS    1.5e-5f
#define SCALE  0.17677669529663687f  // 1/sqrt(32)

// ---- WMMA wrapper (D = A[16x32 f16] * B[32x16 f16] + C[16x16 f32]) --------
__device__ __forceinline__ v8f wmma16(v16h a, v16h b, v8f c) {
  return __builtin_amdgcn_wmma_f32_16x16x32_f16(false, a, false, b,
                                                (short)0, c, false, false);
}

// ---- Operand loaders following ISA 7.12.2 lane layouts --------------------
// A-matrix 16x32 f16: lane l<16 -> row l, K = {0..7, 16..23};
//                     lane l>=16 -> row l-16, K = {8..15, 24..31}
template <typename P>
__device__ __forceinline__ v16h load_A16(P p, int ld, int lane) {
  int row = lane & 15;
  int kh  = (lane >> 4) * 8;
  P r0 = p + (size_t)row * ld + kh;
  v8h lo = *(const v8h*)(r0);
  v8h hi = *(const v8h*)(r0 + 16);
  return __builtin_shufflevector(lo, hi, 0, 1, 2, 3, 4, 5, 6, 7,
                                         8, 9, 10, 11, 12, 13, 14, 15);
}

// B-matrix 32x16 f16: lane l<16 -> col l, K = {0..7, 16..23}; lane l>=16 -> col l-16,
// K = {8..15, 24..31}.  Col-major source: element (K,col) at p[col*colStride + K].
template <typename P>
__device__ __forceinline__ v16h load_B_cm(P p, int colStride, int lane) {
  int col = lane & 15;
  int k0  = (lane >> 4) * 8;
  P r0 = p + (size_t)col * colStride + k0;
  v8h lo = *(const v8h*)(r0);
  v8h hi = *(const v8h*)(r0 + 16);
  return __builtin_shufflevector(lo, hi, 0, 1, 2, 3, 4, 5, 6, 7,
                                         8, 9, 10, 11, 12, 13, 14, 15);
}

// ---- Half-wave (16-lane) row max via ds_swizzle xor -----------------------
__device__ __forceinline__ float redmax16(float v) {
  v = fmaxf(v, __int_as_float(__builtin_amdgcn_ds_swizzle(__float_as_int(v), 0x041f)));
  v = fmaxf(v, __int_as_float(__builtin_amdgcn_ds_swizzle(__float_as_int(v), 0x081f)));
  v = fmaxf(v, __int_as_float(__builtin_amdgcn_ds_swizzle(__float_as_int(v), 0x101f)));
  v = fmaxf(v, __int_as_float(__builtin_amdgcn_ds_swizzle(__float_as_int(v), 0x201f)));
  return v;
}

// ---------------------------------------------------------------------------
// Kernel 1: fp32 -> f16 weight conversion
// ---------------------------------------------------------------------------
__global__ __launch_bounds__(256) void cvt_f16(const float* __restrict__ src,
                                               half_t* __restrict__ dst, int n) {
  int i = blockIdx.x * 256 + threadIdx.x;
  if (i < n) dst[i] = (half_t)src[i];
}

// ---------------------------------------------------------------------------
// Kernel 2: channel RMS-norm, fp32 in [B][C][N] -> f16 out TOKEN-MAJOR [B][N][C]
// ---------------------------------------------------------------------------
__global__ __launch_bounds__(256) void rmsnorm_f16(const float* __restrict__ x,
                                                   const float* __restrict__ w,
                                                   half_t* __restrict__ out, int C) {
  int n = blockIdx.x * 256 + threadIdx.x;      // 0..NTOK-1
  int b = blockIdx.y;
  const float* xb = x + (size_t)b * C * NTOK + n;
  float ss = 0.f;
  for (int c = 0; c < C; ++c) {
    float v = xb[(size_t)c * NTOK];
    ss += v * v;
  }
  float rinv = rsqrtf(ss / (float)C + EPS);
  half_t* ob = out + ((size_t)b * NTOK + n) * C;   // token-major row
  v8h buf;
  for (int c = 0; c < C; ++c) {
    float v = xb[(size_t)c * NTOK];
    buf[c & 7] = (half_t)(v * rinv * w[c]);
    if ((c & 7) == 7) *(v8h*)(ob + (c - 7)) = buf;
  }
}

// ---------------------------------------------------------------------------
// Kernel 3: Q / KV projection GEMM.  D[o,n] = sum_c W[o,c]*Xt[n,c] + bias[o]
//   mode 0: Q -> Qdst[b][head][n][hd] (pre-scaled), packed v8h store
//   mode 1: o<128 -> Kdst[b][head][n][hd] packed; o>=128 -> Vdst[b][head][hd][N]
// ---------------------------------------------------------------------------
__global__ __launch_bounds__(256) void gemm_qkv(const half_t* __restrict__ W,
                                                const float* __restrict__ bias,
                                                const half_t* __restrict__ Xt,
                                                half_t* __restrict__ Qdst,
                                                half_t* __restrict__ Kdst,
                                                half_t* __restrict__ Vdst,
                                                int C, int O, int mode, float qscale) {
  int lane = threadIdx.x & 31;
  int gw   = blockIdx.x * 8 + (threadIdx.x >> 5);
  int nt   = gw % NQT;
  int rest = gw / NQT;
  int otl  = O >> 4;
  int ot   = rest % otl;
  int b    = rest / otl;

  const half_t* Wp = W + (size_t)(ot * 16) * C;
  const half_t* Xp = Xt + ((size_t)b * NTOK + nt * 16) * C;

  v8f acc = {};
  for (int c0 = 0; c0 < C; c0 += 32) {
    v16h a  = load_A16(Wp + c0, C, lane);
    v16h bm = load_B_cm(Xp + c0, C, lane);
    acc = wmma16(a, bm, acc);
  }

  int colL = lane & 15;
  int rowA = (lane >> 4) * 8;
  int o0   = ot * 16;
  int n    = nt * 16 + colL;

  if (mode == 0) {
    int head = o0 >> 5;
    int hdl0 = (o0 & 31) + rowA;
    v8h pk;
#pragma unroll
    for (int r = 0; r < 8; ++r)
      pk[r] = (half_t)((acc[r] + bias[o0 + r + rowA]) * qscale);
    *(v8h*)(Qdst + (((size_t)b * HEADS + head) * NTOK + n) * HD + hdl0) = pk;
  } else if (o0 < DIM) {
    int head = o0 >> 5;
    int hdl0 = (o0 & 31) + rowA;
    v8h pk;
#pragma unroll
    for (int r = 0; r < 8; ++r)
      pk[r] = (half_t)(acc[r] + bias[o0 + r + rowA]);
    *(v8h*)(Kdst + (((size_t)b * HEADS + head) * NTOK + n) * HD + hdl0) = pk;
  } else {
    int oo   = o0 - DIM;
    int head = oo >> 5;
    int hdl0 = (oo & 31) + rowA;
    half_t* dst = Vdst + (((size_t)b * HEADS + head) * HD + hdl0) * NTOK + n;
#pragma unroll
    for (int r = 0; r < 8; ++r)
      dst[(size_t)r * NTOK] = (half_t)(acc[r] + bias[DIM + oo + r + rowA]);
  }
}

// ---------------------------------------------------------------------------
// Kernel 4: flash attention. Block = 8 waves sharing one (b,head); each wave
// owns a 16-query tile. K/V 32-token tiles staged through LDS, double-buffered
// (8x cut in L2 traffic). Row sums via an extra WMMA against a ones B-tile.
// ---------------------------------------------------------------------------
__global__ __launch_bounds__(256) void attention(const half_t* __restrict__ Q,
                                                 const half_t* __restrict__ K,
                                                 const half_t* __restrict__ Vt,
                                                 half_t* __restrict__ Oc) {
  // Tile[buf][0..1023]   = K tile  [32 tok][32 d]
  // Tile[buf][1024..2047]= V tile  [32 d][32 tok]
  __shared__ _Float16 Tile[2][2048];   // 8 KB
  __shared__ _Float16 Pl[8 * 512];     // 8 KB : per-wave P bounce

  int tid  = threadIdx.x;
  int lane = tid & 31;
  int wid  = tid >> 5;
  int bh   = blockIdx.y;
  int h    = bh & (HEADS - 1);
  int b    = bh >> 2;
  int qt   = blockIdx.x * 8 + wid;

  const half_t* qb = Q  + (((size_t)b * HEADS + h) * NTOK + qt * 16) * HD;
  const half_t* kb = K  + (((size_t)b * HEADS + h) * NTOK) * HD;
  const half_t* vb = Vt + (((size_t)b * HEADS + h) * HD) * NTOK;  // d-major

  // ---- per-thread cooperative staging assignment (256 thr x 16B = 4KB) ----
  int sub = tid & 127;
  int row = sub >> 2;             // 0..31
  int ch  = (sub & 3) * 8;        // half offset within row
  const half_t* gptr;
  size_t gstep;
  int dsoff;
  if (tid < 128) {                // K tile: [tok][d]
    gptr  = kb + (size_t)row * HD + ch;
    gstep = (size_t)32 * HD;
    dsoff = row * 32 + ch;
  } else {                        // V tile: [d][tok]
    gptr  = vb + (size_t)row * NTOK + ch;
    gstep = 32;
    dsoff = 1024 + row * 32 + ch;
  }

  v16h aq = load_A16(qb, HD, lane);          // Q tile (already * SCALE)
  v16h vones;
#pragma unroll
  for (int e = 0; e < 16; ++e) vones[e] = (half_t)1.0f;

  v8f o0 = {}, o1 = {};
  float m[8], ls[8], al[8];
#pragma unroll
  for (int r = 0; r < 8; ++r) { m[r] = -1e30f; ls[r] = 0.f; }

  half_t* pt = Pl + wid * 512;
  int colL = lane & 15;
  int rowA = (lane >> 4) * 8;

  // prologue: stage tile 0
  *(v8h*)(&Tile[0][dsoff]) = *(const v8h*)gptr;
  gptr += gstep;
  __syncthreads();

  for (int kt = 0; kt < NKT; ++kt) {
    int cur = kt & 1;
    // prefetch next tile into registers (overlaps with compute below)
    v8h st;
    if (kt + 1 < NKT) st = *(const v8h*)gptr;

    const half_t* kt0 = &Tile[cur][0];
    const half_t* vt0 = &Tile[cur][1024];

    // S = Q * K^T  (two 16-col tiles)
    v16h bk0 = load_B_cm(kt0, 32, lane);
    v16h bk1 = load_B_cm(kt0 + 16 * 32, 32, lane);
    v8f z = {};
    v8f s0 = wmma16(aq, bk0, z);
    v8f s1 = wmma16(aq, bk1, z);

    // online softmax: row max + rescale factors (rows live in 16-lane halves)
#pragma unroll
    for (int r = 0; r < 8; ++r) {
      float mx = redmax16(fmaxf(s0[r], s1[r]));
      float mn = fmaxf(m[r], mx);
      al[r] = __expf(m[r] - mn);
      m[r] = mn;
      s0[r] = __expf(s0[r] - mn);
      s1[r] = __expf(s1[r] - mn);
    }

    // D-layout P -> LDS [16 rows][32 cols] f16
#pragma unroll
    for (int r = 0; r < 8; ++r) {
      pt[(r + rowA) * 32 + colL]      = (half_t)s0[r];
      pt[(r + rowA) * 32 + 16 + colL] = (half_t)s1[r];
    }
    asm volatile("s_wait_dscnt 0" ::: "memory");  // wave-local LDS RAW order

    v16h ap = load_A16((const half_t*)pt, 32, lane);  // P as A-operand

    // row sums via WMMA against ones (replicated across columns)
    v8f rs = wmma16(ap, vones, z);

    // PV accumulate
    v16h bv0 = load_B_cm(vt0, 32, lane);              // d = 0..15
    v16h bv1 = load_B_cm(vt0 + 16 * 32, 32, lane);    // d = 16..31
#pragma unroll
    for (int r = 0; r < 8; ++r) {
      ls[r] = ls[r] * al[r] + rs[r];
      o0[r] *= al[r];
      o1[r] *= al[r];
    }
    o0 = wmma16(ap, bv0, o0);
    o1 = wmma16(ap, bv1, o1);

    // stage next tile, then one barrier covers RAW+WAR across double buffer
    if (kt + 1 < NKT) {
      *(v8h*)(&Tile[cur ^ 1][dsoff]) = st;
      gptr += gstep;
    }
    __syncthreads();
  }

  // normalize and write token-major context: Oc[b][n][head*32 + d]
#pragma unroll
  for (int r = 0; r < 8; ++r) {
    float inv = 1.0f / ls[r];
    int n = qt * 16 + r + rowA;
    size_t base = ((size_t)b * NTOK + n) * DIM + h * HD;
    Oc[base + colL]      = (half_t)(o0[r] * inv);
    Oc[base + 16 + colL] = (half_t)(o1[r] * inv);
  }
}

// ---------------------------------------------------------------------------
// Kernel 5: output projection GEMM: out[b][o][n] = sum_d W[o,d]*Oc[b][n][d] + bias
// ---------------------------------------------------------------------------
__global__ __launch_bounds__(256) void proj_gemm(const half_t* __restrict__ W,
                                                 const float* __restrict__ bias,
                                                 const half_t* __restrict__ Xt,
                                                 float* __restrict__ out) {
  int lane = threadIdx.x & 31;
  int gw   = blockIdx.x * 8 + (threadIdx.x >> 5);
  int nt   = gw % NQT;
  int rest = gw / NQT;
  int ot   = rest % (CQ / 16);
  int b    = rest / (CQ / 16);

  const half_t* Wp = W + (size_t)(ot * 16) * DIM;
  const half_t* Xp = Xt + ((size_t)b * NTOK + nt * 16) * DIM;

  v8f acc = {};
#pragma unroll
  for (int d0 = 0; d0 < DIM; d0 += 32) {
    v16h a  = load_A16(Wp + d0, DIM, lane);
    v16h bm = load_B_cm(Xp + d0, DIM, lane);
    acc = wmma16(a, bm, acc);
  }

  int colL = lane & 15;
  int rowA = (lane >> 4) * 8;
#pragma unroll
  for (int r = 0; r < 8; ++r) {
    int o = ot * 16 + r + rowA;
    int n = nt * 16 + colL;
    out[((size_t)b * CQ + o) * NTOK + n] = acc[r] + bias[o];
  }
}

// ---------------------------------------------------------------------------
// Host launcher
// ---------------------------------------------------------------------------
extern "C" void kernel_launch(void* const* d_in, const int* in_sizes, int n_in,
                              void* d_out, int out_size, void* d_ws, size_t ws_size,
                              hipStream_t stream) {
  const float* x         = (const float*)d_in[0];
  const float* y         = (const float*)d_in[1];
  const float* q_norm_w  = (const float*)d_in[2];
  const float* kv_norm_w = (const float*)d_in[3];
  const float* q_w       = (const float*)d_in[4];
  const float* q_b       = (const float*)d_in[5];
  const float* kv_w      = (const float*)d_in[6];
  const float* kv_b      = (const float*)d_in[7];
  const float* proj_w    = (const float*)d_in[8];
  const float* proj_b    = (const float*)d_in[9];
  float* out = (float*)d_out;

  // Workspace carve-out (~23 MB; everything stays L2-resident on MI455X)
  char* ws = (char*)d_ws;
  size_t off = 0;
  auto carve = [&](size_t bytes) -> char* {
    char* p = ws + off;
    off = (off + bytes + 255) & ~(size_t)255;
    return p;
  };
  half_t* qw_h  = (half_t*)carve((size_t)DIM * CQ * 2);
  half_t* kvw_h = (half_t*)carve((size_t)2 * DIM * CKV * 2);
  half_t* pw_h  = (half_t*)carve((size_t)CQ * DIM * 2);
  half_t* xn    = (half_t*)carve((size_t)B_SZ * NTOK * CQ * 2);   // token-major
  half_t* yn    = (half_t*)carve((size_t)B_SZ * NTOK * CKV * 2);  // token-major
  half_t* Qh    = (half_t*)carve((size_t)B_SZ * HEADS * NTOK * HD * 2);
  half_t* Kh    = (half_t*)carve((size_t)B_SZ * HEADS * NTOK * HD * 2);
  half_t* Vt    = (half_t*)carve((size_t)B_SZ * HEADS * HD * NTOK * 2);  // d-major
  half_t* Oc    = (half_t*)carve((size_t)B_SZ * NTOK * DIM * 2);
  (void)ws_size; (void)in_sizes; (void)n_in; (void)out_size;

  // 1) weights -> f16
  cvt_f16<<<(DIM * CQ + 255) / 256, 256, 0, stream>>>(q_w, qw_h, DIM * CQ);
  cvt_f16<<<(2 * DIM * CKV + 255) / 256, 256, 0, stream>>>(kv_w, kvw_h, 2 * DIM * CKV);
  cvt_f16<<<(CQ * DIM + 255) / 256, 256, 0, stream>>>(proj_w, pw_h, CQ * DIM);

  // 2) RMS norms (NTOK = 9 * 256), outputs token-major
  rmsnorm_f16<<<dim3(NTOK / 256, B_SZ), 256, 0, stream>>>(x, q_norm_w, xn, CQ);
  rmsnorm_f16<<<dim3(NTOK / 256, B_SZ), 256, 0, stream>>>(y, kv_norm_w, yn, CKV);

  // 3) Q projection: 4608 tiles -> 576 blocks; KV: 9216 tiles -> 1152 blocks
  gemm_qkv<<<(B_SZ * (DIM / 16) * NQT) / 8, 256, 0, stream>>>(
      qw_h, q_b, xn, Qh, nullptr, nullptr, CQ, DIM, 0, SCALE);
  gemm_qkv<<<(B_SZ * (2 * DIM / 16) * NQT) / 8, 256, 0, stream>>>(
      kvw_h, kv_b, yn, nullptr, Kh, Vt, CKV, 2 * DIM, 1, 1.0f);

  // 4) attention: block = 8 waves sharing one (b,head); grid (18, 16)
  attention<<<dim3(NQT / 8, B_SZ * HEADS), 256, 0, stream>>>(Qh, Kh, Vt, Oc);

  // 5) output projection: 9216 tiles -> 1152 blocks
  proj_gemm<<<(B_SZ * (CQ / 16) * NQT) / 8, 256, 0, stream>>>(pw_h, proj_b, Oc, out);
}
